// TAGNN_80032420594055
// MI455X (gfx1250) — matrix operand order repaired and checked by reading, not compiled, not run on hardware
//
#include <hip/hip_runtime.h>
#include <stdint.h>

#define H 128
#define NGRAPH 64

typedef __attribute__((ext_vector_type(16))) __bf16 v16bf;
typedef __attribute__((ext_vector_type(8)))  float  v8f;

union AFrag { uint4 q[2]; v16bf v; };
union BFrag { uint4 q[2]; v16bf v; };

__device__ __forceinline__ unsigned short f2bf(float f) {
    unsigned u = __float_as_uint(f);
    unsigned r = u + 0x7FFFu + ((u >> 16) & 1u);   // round-to-nearest-even
    return (unsigned short)(r >> 16);
}

// ================= gcn_norm =================
__global__ void deg_kernel(const int* __restrict__ col, const float* __restrict__ ew,
                           float* __restrict__ deg, int E) {
    int t = blockIdx.x * blockDim.x + threadIdx.x;
    if (t >= E) return;
    atomicAdd(&deg[col[t]], ew[t]);
}

__global__ void dinv_kernel(float* __restrict__ deg, int n) {
    int t = blockIdx.x * blockDim.x + threadIdx.x;
    if (t >= n) return;
    float d = deg[t];
    deg[t] = (d > 0.f) ? rsqrtf(d) : 0.f;
}

__global__ void norm_kernel(const int* __restrict__ row, const int* __restrict__ col,
                            const float* __restrict__ ew, const float* __restrict__ dinv,
                            float* __restrict__ norm, int E) {
    int t = blockIdx.x * blockDim.x + threadIdx.x;
    if (t >= E) return;
    norm[t] = dinv[row[t]] * ew[t] * dinv[col[t]];
}

// ================= CSR build (by destination) =================
__global__ void count_kernel(const int* __restrict__ col, int* __restrict__ cnt, int E) {
    int t = blockIdx.x * blockDim.x + threadIdx.x;
    if (t >= E) return;
    atomicAdd(&cnt[col[t]], 1);
}

// inclusive scan of 512-element chunks; chunk sums to bsum
__global__ void scan1_kernel(const int* __restrict__ cnt, int* __restrict__ out,
                             int* __restrict__ bsum, int n) {
    __shared__ int s[512];
    int t = threadIdx.x, base = blockIdx.x * 512;
    int i1 = t, i2 = t + 256;
    s[i1] = (base + i1 < n) ? cnt[base + i1] : 0;
    s[i2] = (base + i2 < n) ? cnt[base + i2] : 0;
    __syncthreads();
    for (int off = 1; off < 512; off <<= 1) {
        int v1 = (i1 >= off) ? s[i1 - off] : 0;
        int v2 = (i2 >= off) ? s[i2 - off] : 0;
        __syncthreads();
        s[i1] += v1; s[i2] += v2;
        __syncthreads();
    }
    if (base + i1 < n) out[base + i1] = s[i1];
    if (base + i2 < n) out[base + i2] = s[i2];
    if (t == 0) bsum[blockIdx.x] = s[511];
}

// single block: bsum -> exclusive scan of chunk sums (nb <= 512)
__global__ void scan2_kernel(int* __restrict__ bsum, int nb) {
    __shared__ int s[512];
    int t = threadIdx.x;
    int i1 = t, i2 = t + 256;
    int o1 = (i1 < nb) ? bsum[i1] : 0;
    int o2 = (i2 < nb) ? bsum[i2] : 0;
    s[i1] = o1; s[i2] = o2;
    __syncthreads();
    for (int off = 1; off < 512; off <<= 1) {
        int v1 = (i1 >= off) ? s[i1 - off] : 0;
        int v2 = (i2 >= off) ? s[i2 - off] : 0;
        __syncthreads();
        s[i1] += v1; s[i2] += v2;
        __syncthreads();
    }
    if (i1 < nb) bsum[i1] = s[i1] - o1;
    if (i2 < nb) bsum[i2] = s[i2] - o2;
}

// rowptr[g] = chunk_inclusive + chunk_offset - cnt = global exclusive scan
__global__ void scan3_kernel(int* __restrict__ rowptr, const int* __restrict__ bsum,
                             const int* __restrict__ cnt, int n) {
    int g = blockIdx.x * blockDim.x + threadIdx.x;
    if (g >= n) return;
    rowptr[g] = rowptr[g] + bsum[g >> 9] - cnt[g];
}

__global__ void fill_kernel(const int* __restrict__ row, const int* __restrict__ col,
                            const float* __restrict__ norm, int* __restrict__ cursor,
                            int* __restrict__ csr_src, float* __restrict__ csr_nrm, int E) {
    int e = blockIdx.x * blockDim.x + threadIdx.x;
    if (e >= E) return;
    int c = col[e];
    int idx = atomicAdd(&cursor[c], 1);
    csr_src[idx] = row[e];
    csr_nrm[idx] = norm[e];
}

// ================= weight repack: fp32 [20][128][128] -> bf16 B-fragments, lane-major =================
// per matrix: idx = (((nt*4 + kb)*32) + lane)*8 + j   (uint32 = bf16 pair k,k+1)
// k = kb*32 + 16*(lane>>4) + 2*j ; n = nt*16 + (lane&15)
__global__ void repack_w_kernel(const float* __restrict__ w, unsigned* __restrict__ wpack,
                                int nmats) {
    int o = blockIdx.x * blockDim.x + threadIdx.x;
    if (o >= nmats * 8192) return;
    int mat  = o >> 13;
    int r    = o & 8191;
    int j    = r & 7;
    int lane = (r >> 3) & 31;
    int kb   = (r >> 8) & 3;
    int nt   = r >> 10;
    int k = kb * 32 + ((lane >> 4) << 4) + 2 * j;
    int n = nt * 16 + (lane & 15);
    const float* m = w + (size_t)mat * H * H;
    unsigned lo = f2bf(m[k * H + n]);
    unsigned hi = f2bf(m[(k + 1) * H + n]);
    wpack[o] = lo | (hi << 16);
}

// ================= fp32 -> bf16 (used once, for x) =================
__global__ void cvt_bf16_kernel(const float* __restrict__ src, unsigned short* __restrict__ dst,
                                int total4) {
    int t = blockIdx.x * blockDim.x + threadIdx.x;
    if (t >= total4) return;
    float4 v = ((const float4*)src)[t];
    unsigned r0 = (unsigned)f2bf(v.x) | ((unsigned)f2bf(v.y) << 16);
    unsigned r1 = (unsigned)f2bf(v.z) | ((unsigned)f2bf(v.w) << 16);
    ((uint2*)dst)[t] = make_uint2(r0, r1);
}

// ================= dense GEMM: acc(+)= hb[N,128](bf16) @ W =================
// block = 256 threads = 8 waves; wave w = col tile; A tile staged in LDS
__global__ void gemm_bf16_kernel(const unsigned short* __restrict__ hb,
                                 const unsigned* __restrict__ wpack,
                                 float* __restrict__ acc, int nrows, int initC) {
    __shared__ unsigned short atile[16][136];   // 272B row pitch
    const int lane = threadIdx.x & 31;
    const int wv   = threadIdx.x >> 5;
    const int mt   = blockIdx.x;
    const int half = lane >> 4;
    const int mn   = lane & 15;
    const bool full = (mt * 16 + 16) <= nrows;   // wave-uniform: scalar branch, no exec churn

    // cooperative A-tile load: 16 rows x 128 bf16, one uint4 per thread
    {
        int r = threadIdx.x >> 4;
        int cidx = (threadIdx.x & 15) * 8;
        int grow = mt * 16 + r;
        if (grow >= nrows) grow = nrows - 1;
        *(uint4*)&atile[r][cidx] = *(const uint4*)(hb + (size_t)grow * H + cidx);
    }
    __syncthreads();

    float* __restrict__ cptr = acc + (size_t)(mt * 16 + half * 8) * H + wv * 16 + mn;

    v8f c;
    if (initC) {
#pragma unroll
        for (int j = 0; j < 8; j++) c[j] = 0.f;
    } else if (full) {
#pragma unroll
        for (int j = 0; j < 8; j++) c[j] = cptr[j * H];
    } else {
#pragma unroll
        for (int j = 0; j < 8; j++) {
            int rr = mt * 16 + half * 8 + j;
            c[j] = (rr < nrows) ? cptr[j * H] : 0.f;
        }
    }

#pragma unroll
    for (int kb = 0; kb < 4; kb++) {
        AFrag a;
        a.q[0] = *(const uint4*)&atile[mn][kb * 32 + half * 16];       // K = k0..k0+7
        a.q[1] = *(const uint4*)&atile[mn][kb * 32 + half * 16 + 8];   // K = k0+8..k0+15
        BFrag b;
        const uint4* wp = (const uint4*)(wpack + (((wv * 4 + kb) * 32) + lane) * 8);
        b.q[0] = wp[0];
        b.q[1] = wp[1];
        c = __builtin_amdgcn_wmma_f32_16x16x32_bf16(false, a.v, false, b.v,
                                                    (short)0, c, false, false);
    }

    if (full) {
#pragma unroll
        for (int j = 0; j < 8; j++) cptr[j * H] = c[j];
    } else {
#pragma unroll
        for (int j = 0; j < 8; j++) {
            int rr = mt * 16 + half * 8 + j;
            if (rr < nrows) cptr[j * H] = c[j];
        }
    }
}

// ================= SpMM (CSR gather, no atomics): one wave per dst node =================
// writes fp32 result + bf16 copy for the following WMMA GEMM
__global__ void spmm_csr_kernel(const int* __restrict__ rowptr, const int* __restrict__ cnt,
                                const int* __restrict__ csr_src, const float* __restrict__ csr_nrm,
                                const float* __restrict__ hin, float* __restrict__ hout,
                                unsigned short* __restrict__ hbout, int n) {
    long long t = (long long)blockIdx.x * blockDim.x + threadIdx.x;
    int v = (int)(t >> 5);
    if (v >= n) return;
    int c = ((int)t & 31) << 2;
    int s0 = rowptr[v];
    int e1 = s0 + cnt[v];
    float4 acc = make_float4(0.f, 0.f, 0.f, 0.f);
    for (int e = s0; e < e1; e++) {
        float w = csr_nrm[e];
        int u = csr_src[e];
        float4 hv = *(const float4*)(hin + (size_t)u * H + c);
        acc.x += w * hv.x; acc.y += w * hv.y;
        acc.z += w * hv.z; acc.w += w * hv.w;
    }
    *(float4*)(hout + (size_t)v * H + c) = acc;
    unsigned r0 = (unsigned)f2bf(acc.x) | ((unsigned)f2bf(acc.y) << 16);
    unsigned r1 = (unsigned)f2bf(acc.z) | ((unsigned)f2bf(acc.w) << 16);
    *(uint2*)(hbout + (size_t)v * H + c) = make_uint2(r0, r1);
}

// ================= bias + relu (+ bf16 copy for next layer's GEMM) =================
__global__ void bias_relu_kernel(const float* __restrict__ acc, const float* __restrict__ bias,
                                 float* __restrict__ out, unsigned short* __restrict__ hbout,
                                 int total4) {
    int t = blockIdx.x * blockDim.x + threadIdx.x;
    if (t >= total4) return;
    int cb = (t & 31) << 2;
    float4 v = ((const float4*)acc)[t];
    v.x = fmaxf(v.x + bias[cb + 0], 0.f);
    v.y = fmaxf(v.y + bias[cb + 1], 0.f);
    v.z = fmaxf(v.z + bias[cb + 2], 0.f);
    v.w = fmaxf(v.w + bias[cb + 3], 0.f);
    ((float4*)out)[t] = v;
    unsigned r0 = (unsigned)f2bf(v.x) | ((unsigned)f2bf(v.y) << 16);
    unsigned r1 = (unsigned)f2bf(v.z) | ((unsigned)f2bf(v.w) << 16);
    ((uint2*)hbout)[t] = make_uint2(r0, r1);
}

// ================= global mean pool =================
__global__ void pool_kernel(const float* __restrict__ h, const int* __restrict__ batch,
                            float* __restrict__ sums, float* __restrict__ cnt, int n) {
    long long t = (long long)blockIdx.x * blockDim.x + threadIdx.x;
    int i = (int)(t >> 5);
    if (i >= n) return;
    int c = ((int)t & 31) << 2;
    int g = batch[i];
    float4 v = *(const float4*)(h + (size_t)i * H + c);
    float* dst = sums + g * H + c;
    atomicAdd(dst + 0, v.x);
    atomicAdd(dst + 1, v.y);
    atomicAdd(dst + 2, v.z);
    atomicAdd(dst + 3, v.w);
    if (c == 0) atomicAdd(&cnt[g], 1.0f);
}

// ================= MLP head =================
__global__ void mlp_kernel(const float* __restrict__ sums, const float* __restrict__ cnt,
                           const float* __restrict__ w1, const float* __restrict__ b1,
                           const float* __restrict__ w2, const float* __restrict__ b2,
                           float* __restrict__ out) {
    __shared__ float mean[H];
    __shared__ float red[H];
    int t = threadIdx.x;  // 128 threads
    for (int g = 0; g < NGRAPH; g++) {
        float inv = 1.0f / fmaxf(cnt[g], 1.0f);
        mean[t] = sums[g * H + t] * inv;
        __syncthreads();
        float v = b1[t];
#pragma unroll 8
        for (int i = 0; i < H; i++) v += mean[i] * w1[i * H + t];
        red[t] = v * w2[t];
        __syncthreads();
        for (int s = 64; s > 0; s >>= 1) {
            if (t < s) red[t] += red[t + s];
            __syncthreads();
        }
        if (t == 0) out[g] = 1.0f / (1.0f + expf(-(red[0] + b2[0])));
        __syncthreads();
    }
}

extern "C" void kernel_launch(void* const* d_in, const int* in_sizes, int n_in,
                              void* d_out, int out_size, void* d_ws, size_t ws_size,
                              hipStream_t stream) {
    const float* x       = (const float*)d_in[0];
    const int*   ei      = (const int*)d_in[1];
    const float* ew      = (const float*)d_in[2];
    const int*   batch   = (const int*)d_in[3];
    const float* conv_ws = (const float*)d_in[4];
    const float* conv_bs = (const float*)d_in[5];
    const float* lin1_w  = (const float*)d_in[6];
    const float* lin1_b  = (const float*)d_in[7];
    const float* lin2_w  = (const float*)d_in[8];
    const float* lin2_b  = (const float*)d_in[9];

    const int N = in_sizes[0] / H;
    const int E = in_sizes[2];
    const int* row = ei;
    const int* col = ei + E;

    size_t off = 0;
    auto carve = [&](size_t bytes) -> char* {
        char* p = (char*)d_ws + off;
        off += (bytes + 255) & ~(size_t)255;
        return p;
    };
    const size_t hbytes = (size_t)N * H * sizeof(float);
    float*          hA      = (float*)carve(hbytes);
    float*          hB      = (float*)carve(hbytes);
    float*          acc     = (float*)carve(hbytes);
    unsigned short* hb      = (unsigned short*)carve((size_t)N * H * sizeof(unsigned short));
    unsigned*       wpack   = (unsigned*)carve((size_t)20 * 8192 * sizeof(unsigned));
    float*          norm    = (float*)carve((size_t)E * sizeof(float));
    float*          deg     = (float*)carve((size_t)N * sizeof(float));
    int*            cnt     = (int*)carve((size_t)N * sizeof(int));
    int*            rowptr  = (int*)carve((size_t)N * sizeof(int));
    int*            cursor  = (int*)carve((size_t)N * sizeof(int));
    int*            bsum    = (int*)carve(512 * sizeof(int));
    int*            csr_src = (int*)carve((size_t)E * sizeof(int));
    float*          csr_nrm = (float*)carve((size_t)E * sizeof(float));
    float*          psum    = (float*)carve((size_t)NGRAPH * H * sizeof(float));
    float*          pcnt    = (float*)carve((size_t)NGRAPH * sizeof(float));

    const int BT = 256;
    auto blocks = [&](long long n) { return (unsigned)((n + BT - 1) / BT); };
    const int total4 = N * H / 4;
    const int nb1 = (N + 511) / 512;

    // 1) gcn_norm
    hipMemsetAsync(deg, 0, (size_t)N * sizeof(float), stream);
    deg_kernel<<<blocks(E), BT, 0, stream>>>(col, ew, deg, E);
    dinv_kernel<<<blocks(N), BT, 0, stream>>>(deg, N);
    norm_kernel<<<blocks(E), BT, 0, stream>>>(row, col, ew, deg, norm, E);

    // 2) CSR-by-destination (built once, reused by all 15 propagations; no steady-state atomics)
    hipMemsetAsync(cnt, 0, (size_t)N * sizeof(int), stream);
    count_kernel<<<blocks(E), BT, 0, stream>>>(col, cnt, E);
    scan1_kernel<<<nb1, BT, 0, stream>>>(cnt, rowptr, bsum, N);
    scan2_kernel<<<1, BT, 0, stream>>>(bsum, nb1);
    scan3_kernel<<<blocks(N), BT, 0, stream>>>(rowptr, bsum, cnt, N);
    hipMemcpyAsync(cursor, rowptr, (size_t)N * sizeof(int), hipMemcpyDeviceToDevice, stream);
    fill_kernel<<<blocks(E), BT, 0, stream>>>(row, col, norm, cursor, csr_src, csr_nrm, E);

    // 3) weight repack to WMMA B-fragment order (lane-major -> b128 loads)
    repack_w_kernel<<<blocks(20 * 8192), BT, 0, stream>>>(conv_ws, wpack, 20);

    // 4) h = x (fp32 working copy + bf16 copy)
    hipMemcpyAsync(hA, x, hbytes, hipMemcpyDeviceToDevice, stream);
    cvt_bf16_kernel<<<blocks(total4), BT, 0, stream>>>(x, hb, total4);
    float* cur = hA;
    float* tmp = hB;

    const int rowTiles = (N + 15) / 16;
    const long long nodeThreads = (long long)N * 32;

    for (int layer = 0; layer < 5; layer++) {
        // acc = h @ W0
        gemm_bf16_kernel<<<rowTiles, BT, 0, stream>>>(hb, wpack + (layer * 4 + 0) * 8192, acc, N, 1);
        // k = 1..3: h <- A_hat h (gather, no atomics); acc += h @ Wk
        for (int k = 1; k <= 3; k++) {
            spmm_csr_kernel<<<blocks(nodeThreads), BT, 0, stream>>>(rowptr, cnt, csr_src, csr_nrm,
                                                                    cur, tmp, hb, N);
            float* sw = cur; cur = tmp; tmp = sw;
            gemm_bf16_kernel<<<rowTiles, BT, 0, stream>>>(hb, wpack + (layer * 4 + k) * 8192, acc, N, 0);
        }
        // h_next = relu(acc + b) -> fp32 + bf16
        bias_relu_kernel<<<blocks(total4), BT, 0, stream>>>(acc, conv_bs + layer * H, cur, hb, total4);
    }

    // 5) mean pool + MLP head
    hipMemsetAsync(psum, 0, (size_t)NGRAPH * H * sizeof(float), stream);
    hipMemsetAsync(pcnt, 0, (size_t)NGRAPH * sizeof(float), stream);
    pool_kernel<<<blocks(nodeThreads), BT, 0, stream>>>(cur, batch, psum, pcnt, N);
    mlp_kernel<<<1, H, 0, stream>>>(psum, pcnt, lin1_w, lin1_b, lin2_w, lin2_b, (float*)d_out);
}
